// Generator_10642928959685
// MI455X (gfx1250) — matrix operand order repaired and checked by reading, not compiled, not run on hardware
//
#include <hip/hip_runtime.h>
#include <cstddef>

// ---------------------------------------------------------------------------
// MI455X (gfx1250) generator forward pass.
// All linears run as wave32 WMMA GEMMs (v_wmma_f32_16x16x32_f16, f32 acc).
// BatchNorms are folded into fp16 weights/biases at launch; weights are stored
// in WMMA *fragment order* so B fragments are single contiguous 32B loads per
// lane. A tiles are double-buffered in LDS (one barrier per k-step); B
// fragments are register-double-buffered so their global-load latency hides
// under the WMMAs. Residual add + post-residual BN + activation are fused
// into the GEMM epilogue. The tiny 11x11 attention core is a per-batch LDS
// kernel.
// ---------------------------------------------------------------------------

typedef __attribute__((ext_vector_type(16))) _Float16 v16h;
typedef __attribute__((ext_vector_type(8)))  float    v8f;

#define GEMM_M 11264   // B * OBJ = 1024 * 11
#define TILE_M 128
#define TILE_N 64

// B fragment-ordered weight layout:
//   tile (kt = k/32, nt = n/16) is 512 halves: lane*16 + elem, with
//   lane = ((k>>4)&1)*16 + (n&15), elem = k&15.
// Each lane's fragment = 32 contiguous bytes -> 2x global_load_b128.

// ---------------------------------------------------------------------------
// Generic fp16 WMMA GEMM:
//   out[m,n] = act( s2[n]*(A@Wt + bias[n] + resid[m,n]) + b2[n] )
//   A     : [M, K] fp16 row-major (K multiple of 32)
//   Wt    : fragment-ordered fp16, padded to Nfrag (multiple of 64)
//   Np    : logical output width (out/resid stride), act: 0 none,1 relu,2 sigmoid
// Block = 256 threads = 8 waves; block tile 128x64; wave tile 32x32 (4 WMMA).
// ---------------------------------------------------------------------------
__global__ __launch_bounds__(256)
void wmma_gemm(const _Float16* __restrict__ A, int K,
               const _Float16* __restrict__ Wt, int Nfrag, int Np,
               const float* __restrict__ bias,
               const float* __restrict__ s2, const float* __restrict__ b2,
               const _Float16* __restrict__ resid,
               _Float16* __restrict__ out, float* __restrict__ out32,
               int act)
{
    __shared__ _Float16 Als[2][TILE_M][32];   // double-buffered A tile, m-major

    const int tid   = threadIdx.x;
    const int wid   = tid >> 5;
    const int lane  = tid & 31;
    const int half  = lane >> 4;     // lane group 0/1
    const int l15   = lane & 15;
    const int m0    = blockIdx.x * TILE_M;
    const int n0    = blockIdx.y * TILE_N;
    const int msub  = wid & 3;       // which 32-row slab (0..3)
    const int nhalf = wid >> 2;      // which 32-col slab (0..1)

    // wave's two 16-col fragment tiles (in units of 16 columns)
    const int nt0 = ((n0 + (nhalf << 5)) >> 4);
    const size_t kstride = (size_t)(Nfrag >> 4) * 512;   // halves per k-step
    const _Float16* bptr = Wt + (size_t)nt0 * 512 + lane * 16;

    // A staging addresses for this thread
    const int ar = tid >> 1;              // 0..127
    const int ac = (tid & 1) << 4;        // 0 or 16
    const _Float16* arow = A + (size_t)(m0 + ar) * K + ac;

    v8f acc[2][2] = {};

    // prologue: stage tile 0 into buffer 0, preload B fragments for step 0
    {
        *(float4*)(&Als[0][ar][ac])     = *(const float4*)(arow);
        *(float4*)(&Als[0][ar][ac + 8]) = *(const float4*)(arow + 8);
    }
    v16h bf0 = *(const v16h*)(bptr);
    v16h bf1 = *(const v16h*)(bptr + 512);
    __syncthreads();

    int p = 0;
    int kt = 0;
    for (; kt + 32 < K; kt += 32) {
        // --- A fragments for the wave's two 16-row slabs (buffer p) ---
        v16h af[2];
#pragma unroll
        for (int mi = 0; mi < 2; ++mi) {
            const int mrow = (msub << 5) + (mi << 4) + l15;
            *((float4*)&af[mi])     = *(const float4*)&Als[p][mrow][half << 4];
            *((float4*)&af[mi] + 1) = *(const float4*)&Als[p][mrow][(half << 4) + 8];
        }

        // --- unconditional: stage next A tile, preload next B fragments ---
        {
            const _Float16* ap = arow + kt + 32;
            *(float4*)(&Als[p ^ 1][ar][ac])     = *(const float4*)(ap);
            *(float4*)(&Als[p ^ 1][ar][ac + 8]) = *(const float4*)(ap + 8);
            __builtin_prefetch(ap + 32, 0, 1);   // speculative, safe past end
        }
        const _Float16* bp = bptr + (size_t)((kt >> 5) + 1) * kstride;
        const v16h nbf0 = *(const v16h*)(bp);
        const v16h nbf1 = *(const v16h*)(bp + 512);

        acc[0][0] = __builtin_amdgcn_wmma_f32_16x16x32_f16(
            false, af[0], false, bf0, (short)0, acc[0][0], false, false);
        acc[0][1] = __builtin_amdgcn_wmma_f32_16x16x32_f16(
            false, af[0], false, bf1, (short)0, acc[0][1], false, false);
        acc[1][0] = __builtin_amdgcn_wmma_f32_16x16x32_f16(
            false, af[1], false, bf0, (short)0, acc[1][0], false, false);
        acc[1][1] = __builtin_amdgcn_wmma_f32_16x16x32_f16(
            false, af[1], false, bf1, (short)0, acc[1][1], false, false);

        __syncthreads();   // single barrier per k-step (double buffered)
        bf0 = nbf0;
        bf1 = nbf1;
        p ^= 1;
    }

    // --- tail k-step (no staging, no barrier) ---
    {
        v16h af[2];
#pragma unroll
        for (int mi = 0; mi < 2; ++mi) {
            const int mrow = (msub << 5) + (mi << 4) + l15;
            *((float4*)&af[mi])     = *(const float4*)&Als[p][mrow][half << 4];
            *((float4*)&af[mi] + 1) = *(const float4*)&Als[p][mrow][(half << 4) + 8];
        }
        acc[0][0] = __builtin_amdgcn_wmma_f32_16x16x32_f16(
            false, af[0], false, bf0, (short)0, acc[0][0], false, false);
        acc[0][1] = __builtin_amdgcn_wmma_f32_16x16x32_f16(
            false, af[0], false, bf1, (short)0, acc[0][1], false, false);
        acc[1][0] = __builtin_amdgcn_wmma_f32_16x16x32_f16(
            false, af[1], false, bf0, (short)0, acc[1][0], false, false);
        acc[1][1] = __builtin_amdgcn_wmma_f32_16x16x32_f16(
            false, af[1], false, bf1, (short)0, acc[1][1], false, false);
    }

    // --- epilogue: bias, residual, per-channel scale/shift, activation ---
#pragma unroll
    for (int g = 0; g < 2; ++g) {
        const int col = n0 + (nhalf << 5) + (g << 4) + l15;
        if (col < Np) {
            const float bi = bias ? bias[col] : 0.f;
            const float sc = s2 ? s2[col] : 1.f;
            const float sh = b2 ? b2[col] : 0.f;
#pragma unroll
            for (int mi = 0; mi < 2; ++mi) {
                const int row_base = m0 + (msub << 5) + (mi << 4) + (half << 3);
#pragma unroll
                for (int i = 0; i < 8; ++i) {
                    const int row = row_base + i;
                    float v = acc[mi][g][i] + bi;
                    if (resid) v += (float)resid[(size_t)row * Np + col];
                    v = v * sc + sh;
                    if (act == 1)      v = v > 0.f ? v : 0.f;
                    else if (act == 2) v = 1.f / (1.f + expf(-v));
                    out[(size_t)row * Np + col] = (_Float16)v;
                    if (out32) out32[(size_t)row * Np + col] = v;
                }
            }
        }
    }
}

// ---------------------------------------------------------------------------
// Attention core: per batch b (block), S = 11 positions:
//   w[i][j] = (1/11) * sum_c k[b,i,c] * q[b,j,c]
//   fr[j,c] = sum_i w[i][j] * v[b,i,c]          (written in place over v)
// ---------------------------------------------------------------------------
__global__ __launch_bounds__(128)
void attn_core(const _Float16* __restrict__ kmat,
               const _Float16* __restrict__ qmat,
               _Float16* __restrict__ vmat, int C)
{
    __shared__ _Float16 vls[11 * 1024];
    __shared__ float    wls[128];

    const int b = blockIdx.x;
    const int t = threadIdx.x;
    const size_t base = (size_t)b * 11 * C;

    for (int idx = t; idx < 11 * C; idx += 128)
        vls[idx] = vmat[base + idx];

    if (t < 121) {
        const int i = t / 11, j = t % 11;
        float acc = 0.f;
        for (int c = 0; c < C; ++c)
            acc += (float)kmat[base + i * C + c] * (float)qmat[base + j * C + c];
        wls[t] = acc * (1.f / 11.f);
    }
    __syncthreads();

    for (int idx = t; idx < 11 * C; idx += 128) {
        const int j = idx / C, c = idx % C;
        float s = 0.f;
#pragma unroll
        for (int i = 0; i < 11; ++i)
            s += wls[i * 11 + j] * (float)vls[i * C + c];
        vmat[base + idx] = (_Float16)s;
    }
}

// ---------------------------------------------------------------------------
// Prep kernels
// ---------------------------------------------------------------------------
__global__ void fold_lin(const float* __restrict__ W, const float* __restrict__ bsrc,
                         const float* __restrict__ gamma, const float* __restrict__ beta,
                         const float* __restrict__ mean, const float* __restrict__ var,
                         _Float16* __restrict__ Wt, float* __restrict__ bias,
                         int cin, int cout, int Nfrag, int noff)
{
    const int idx = blockIdx.x * blockDim.x + threadIdx.x;
    if (idx >= cin * cout) return;
    const int nsrc = idx / cin;
    const int k    = idx % cin;
    const int n    = noff + nsrc;
    float sc = 1.f, sh = 0.f;
    if (gamma) {
        sc = gamma[nsrc] * rsqrtf(var[nsrc] + 1e-5f);
        sh = beta[nsrc] - mean[nsrc] * sc;
    }
    // fragment-ordered destination index
    const size_t dst = ((size_t)(k >> 5) * (Nfrag >> 4) + (n >> 4)) * 512
                     + (size_t)(((k >> 4) & 1) << 8) + ((n & 15) << 4) + (k & 15);
    Wt[dst] = (_Float16)(sc * W[(size_t)nsrc * cin + k]);
    if (k == 0) bias[n] = sc * bsrc[nsrc] + sh;
}

__global__ void bn2ss(const float* __restrict__ g, const float* __restrict__ be,
                      const float* __restrict__ me, const float* __restrict__ va,
                      float* __restrict__ s2, float* __restrict__ b2, int n)
{
    const int i = blockIdx.x * blockDim.x + threadIdx.x;
    if (i >= n) return;
    const float sc = g[i] * rsqrtf(va[i] + 1e-5f);
    s2[i] = sc;
    b2[i] = be[i] - me[i] * sc;
}

__global__ void zero_h(_Float16* p, size_t n)
{
    const size_t i = blockIdx.x * (size_t)blockDim.x + threadIdx.x;
    if (i < n) p[i] = (_Float16)0.f;
}

__global__ void zero_f(float* p, size_t n)
{
    const size_t i = blockIdx.x * (size_t)blockDim.x + threadIdx.x;
    if (i < n) p[i] = 0.f;
}

__global__ void pad_input(const float* __restrict__ noise, _Float16* __restrict__ x, int M)
{
    const int t = blockIdx.x * blockDim.x + threadIdx.x;
    if (t >= M * 32) return;
    const int m = t >> 5, k = t & 31;
    x[t] = (k < 11) ? (_Float16)noise[m * 11 + k] : (_Float16)0.f;
}

__global__ void head_mask(const float* __restrict__ raw, float* __restrict__ out, int M)
{
    const int m = blockIdx.x * blockDim.x + threadIdx.x;
    if (m >= M) return;
    const float* r = raw + (size_t)m * 16;
    int cnt = 0;
    for (int j = 4; j < 11; ++j)
        if (r[j] > 0.5f) cnt++;
    const float mask = (cnt >= 1) ? 1.f : 0.f;
    for (int j = 0; j < 11; ++j)
        out[(size_t)m * 11 + j] = r[j] * mask;
}

// ---------------------------------------------------------------------------
// Host orchestration
// ---------------------------------------------------------------------------
extern "C" void kernel_launch(void* const* d_in, const int* in_sizes, int n_in,
                              void* d_out, int out_size, void* d_ws, size_t ws_size,
                              hipStream_t stream)
{
    (void)in_sizes; (void)n_in; (void)out_size; (void)ws_size;
    const int M = GEMM_M;
    auto F = [&](int i) { return (const float*)d_in[i]; };

    // bump allocator over the workspace (256B aligned)
    char* base = (char*)d_ws;
    size_t off = 0;
    auto alloc = [&](size_t bytes) -> char* {
        char* p = base + off;
        off += (bytes + 255) & ~(size_t)255;
        return p;
    };
    auto a16 = [&](size_t e) { return (_Float16*)alloc(e * 2); };
    auto a32 = [&](size_t e) { return (float*)alloc(e * 4); };

    // ---- folded fp16 weight arena (fragment-ordered, zeroed each launch) ----
    const size_t w_begin = off;
    _Float16* w_h00  = a16(32 * 256);
    _Float16* w_h01  = a16(32 * 64);
    _Float16* w_h03a = a16(64 * 64);
    _Float16* w_h03b = a16(64 * 256);
    _Float16* w_h10  = a16(256 * 1024);
    _Float16* w_h11  = a16(1024 * 256);
    _Float16* w_h13a = a16(256 * 256);
    _Float16* w_h13b = a16(256 * 1024);
    _Float16 *waV[4], *waK[4], *waQ[4], *waR[4];
    const int AC[4] = {256, 256, 1024, 1024};
    for (int s = 0; s < 4; ++s) {
        const size_t e = (size_t)AC[s] * AC[s];
        waV[s] = a16(e); waK[s] = a16(e); waQ[s] = a16(e); waR[s] = a16(e);
    }
    _Float16* w_head = a16(1024 * 64);   // Nfrag = 64 (cols 16..63 stay zero)
    const size_t w_end = off;

    // ---- f32 bias arena ----
    const size_t b_begin = off;
    float* bi_h00  = a32(256);
    float* bi_h01  = a32(64);
    float* bi_h03a = a32(64);
    float* bi_h03b = a32(256);
    float* bi_h10  = a32(1024);
    float* bi_h11  = a32(256);
    float* bi_h13a = a32(256);
    float* bi_h13b = a32(1024);
    float *biV[4], *biK[4], *biQ[4], *biR[4];
    for (int s = 0; s < 4; ++s) {
        biV[s] = a32(AC[s]); biK[s] = a32(AC[s]);
        biQ[s] = a32(AC[s]); biR[s] = a32(AC[s]);
    }
    float* bi_head = a32(16);
    const size_t b_end = off;

    // ---- post-residual BN scale/shift (bn_x0r..bn_x3r) ----
    float *ss[4], *sb[4];
    for (int s = 0; s < 4; ++s) { ss[s] = a32(AC[s]); sb[s] = a32(AC[s]); }

    // ---- fp16 activations ----
    _Float16* x  = a16((size_t)M * 32);    // padded input (reused as head fp16 sink)
    _Float16* t0 = a16((size_t)M * 256);   // h0_0
    _Float16* t1 = a16((size_t)M * 64);    // h0_1
    _Float16* t2 = a16((size_t)M * 64);    // h0_3a
    _Float16* g0 = a16((size_t)M * 256);   // g (stage 0)
    _Float16* u1 = a16((size_t)M * 256);   // h1_1
    _Float16* u2 = a16((size_t)M * 256);   // h1_3a
    _Float16* G  = a16((size_t)M * 1024);  // h1_0 / g (stage 1), residual in place
    _Float16* vb = a16((size_t)M * 1024);  // v / fr
    _Float16* kb = a16((size_t)M * 1024);
    _Float16* qb = a16((size_t)M * 1024);
    float*    raw = a32((size_t)M * 16);   // f32 head output (sigmoid applied)

    // ---- zero arenas (covers all K/N zero padding in fragment layout) ----
    {
        const size_t n = (w_end - w_begin) / 2;
        zero_h<<<(int)((n + 255) / 256), 256, 0, stream>>>((_Float16*)(base + w_begin), n);
    }
    {
        const size_t n = (b_end - b_begin) / 4;
        zero_f<<<(int)((n + 255) / 256), 256, 0, stream>>>((float*)(base + b_begin), n);
    }

    // ---- fold weights (+BN) into fragment-ordered fp16 Wt / f32 bias ----
    auto fold = [&](int wi, int bi, int bn, _Float16* Wt, float* bias,
                    int cin, int cout, int Nfrag, int noff) {
        const int total = cin * cout;
        const float *g = nullptr, *be = nullptr, *me = nullptr, *va = nullptr;
        if (bn >= 0) { g = F(bn); be = F(bn + 1); me = F(bn + 2); va = F(bn + 3); }
        fold_lin<<<(total + 255) / 256, 256, 0, stream>>>(
            F(wi), F(bi), g, be, me, va, Wt, bias, cin, cout, Nfrag, noff);
    };

    // input index map (dict flattening order of the reference params)
    fold(1, 2, 3,     w_h00,  bi_h00,  11,   256,  256,  0);
    fold(7, 8, 9,     w_h01,  bi_h01,  11,   64,   64,   0);
    fold(13, 14, 15,  w_h03a, bi_h03a, 64,   64,   64,   0);
    fold(19, 20, 21,  w_h03b, bi_h03b, 64,   256,  256,  0);
    fold(25, 26, 27,  w_h10,  bi_h10,  256,  1024, 1024, 0);
    fold(31, 32, 33,  w_h11,  bi_h11,  1024, 256,  256,  0);
    fold(37, 38, 39,  w_h13a, bi_h13a, 256,  256,  256,  0);
    fold(43, 44, 45,  w_h13b, bi_h13b, 256,  1024, 1024, 0);

    const int attnW[4] = {49, 57, 65, 73};
    const int bnx[4]   = {81, 89, 97, 105};    // bn_x0..bn_x3 (folded into r-linear)
    const int bnxr[4]  = {85, 93, 101, 109};   // bn_x0r..bn_x3r (epilogue scale/shift)
    for (int s = 0; s < 4; ++s) {
        const int C = AC[s], wb = attnW[s];
        fold(wb + 0, wb + 1, -1,      waV[s], biV[s], C, C, C, 0);
        fold(wb + 2, wb + 3, -1,      waK[s], biK[s], C, C, C, 0);
        fold(wb + 4, wb + 5, -1,      waQ[s], biQ[s], C, C, C, 0);
        fold(wb + 6, wb + 7, bnx[s],  waR[s], biR[s], C, C, C, 0);
        bn2ss<<<(C + 255) / 256, 256, 0, stream>>>(
            F(bnxr[s]), F(bnxr[s] + 1), F(bnxr[s] + 2), F(bnxr[s] + 3),
            ss[s], sb[s], C);
    }
    fold(113, 114, -1, w_head, bi_head, 1024, 4, 64, 0);  // bbox -> cols 0..3
    fold(115, 116, -1, w_head, bi_head, 1024, 7, 64, 4);  // cls  -> cols 4..10

    // ---- input padding: [M,11] f32 -> [M,32] fp16 ----
    pad_input<<<(M * 32 + 255) / 256, 256, 0, stream>>>(F(0), x, M);

    // ---- GEMM chain ----
    auto gemm = [&](const _Float16* A, int K, const _Float16* Wt, int Nfrag, int Np,
                    const float* bias, const float* sc, const float* sh,
                    const _Float16* resid, _Float16* out, float* out32, int act) {
        dim3 grid(M / TILE_M, Nfrag / TILE_N);
        wmma_gemm<<<grid, 256, 0, stream>>>(A, K, Wt, Nfrag, Np, bias, sc, sh,
                                            resid, out, out32, act);
    };

    gemm(x,  32, w_h00,  256, 256, bi_h00,  nullptr, nullptr, nullptr, t0, nullptr, 0);
    gemm(x,  32, w_h01,  64,  64,  bi_h01,  nullptr, nullptr, nullptr, t1, nullptr, 1);
    gemm(t1, 64, w_h03a, 64,  64,  bi_h03a, nullptr, nullptr, nullptr, t2, nullptr, 1);
    gemm(t2, 64, w_h03b, 256, 256, bi_h03b, nullptr, nullptr, t0,      g0, nullptr, 1);

    for (int s = 0; s < 2; ++s) {  // attn0, attn1 (C = 256)
        gemm(g0, 256, waV[s], 256, 256, biV[s], nullptr, nullptr, nullptr, vb, nullptr, 0);
        gemm(g0, 256, waK[s], 256, 256, biK[s], nullptr, nullptr, nullptr, kb, nullptr, 0);
        gemm(g0, 256, waQ[s], 256, 256, biQ[s], nullptr, nullptr, nullptr, qb, nullptr, 0);
        attn_core<<<1024, 128, 0, stream>>>(kb, qb, vb, 256);
        gemm(vb, 256, waR[s], 256, 256, biR[s], ss[s], sb[s], g0, g0, nullptr, 1);
    }

    gemm(g0, 256,  w_h10,  1024, 1024, bi_h10,  nullptr, nullptr, nullptr, G,  nullptr, 0);
    gemm(G,  1024, w_h11,  256,  256,  bi_h11,  nullptr, nullptr, nullptr, u1, nullptr, 1);
    gemm(u1, 256,  w_h13a, 256,  256,  bi_h13a, nullptr, nullptr, nullptr, u2, nullptr, 1);
    gemm(u2, 256,  w_h13b, 1024, 1024, bi_h13b, nullptr, nullptr, G,       G,  nullptr, 1);

    for (int s = 2; s < 4; ++s) {  // attn2, attn3 (C = 1024)
        gemm(G, 1024, waV[s], 1024, 1024, biV[s], nullptr, nullptr, nullptr, vb, nullptr, 0);
        gemm(G, 1024, waK[s], 1024, 1024, biK[s], nullptr, nullptr, nullptr, kb, nullptr, 0);
        gemm(G, 1024, waQ[s], 1024, 1024, biQ[s], nullptr, nullptr, nullptr, qb, nullptr, 0);
        attn_core<<<1024, 128, 0, stream>>>(kb, qb, vb, 1024);
        gemm(vb, 1024, waR[s], 1024, 1024, biR[s], ss[s], sb[s], G, G, nullptr, 1);
    }

    // head: [M,1024] -> [M,16] logical (Nfrag=64) with sigmoid, f32 side output
    gemm(G, 1024, w_head, 64, 16, bi_head, nullptr, nullptr, nullptr, x, raw, 2);

    // mask + final output [M, 11]
    head_mask<<<(M + 255) / 256, 256, 0, stream>>>(raw, (float*)d_out, M);
}
